// MogLSTM_22531398435021
// MI455X (gfx1250) — compile-verified, matrix operand order checked by reading, not run on hardware
//
#include <hip/hip_runtime.h>
#include <hip/hip_bf16.h>
#include <stdint.h>

typedef __bf16 bf16;
typedef __attribute__((ext_vector_type(16))) __bf16 v16bf;
typedef __attribute__((ext_vector_type(8)))  float  v8f;

#define B_   128
#define S_   512
#define D_   512
#define H_   512
#define NBLK 32
#define TPB  256

// ---------- helpers ----------

__device__ __forceinline__ bf16 to_bf16(float f) {
  union { float f; unsigned u; } v; v.f = f;
  unsigned r = v.u + 0x7FFFu + ((v.u >> 16) & 1u);   // round-to-nearest-even
  unsigned short h = (unsigned short)(r >> 16);
  union { unsigned short s; bf16 b; } o; o.s = h;
  return o.b;
}

__device__ __forceinline__ float sigmoid_(float x) {
  return 1.0f / (1.0f + __expf(-x));
}
__device__ __forceinline__ float tanh_(float x) {
  float e = __expf(2.0f * x);
  return 1.0f - 2.0f / (e + 1.0f);   // saturates cleanly to +/-1
}

// Load one 16-bit A/B fragment (16x32 tile, this lane's 16 bf16 values):
// two contiguous 16-byte chunks at +0 and +16 elements.
__device__ __forceinline__ v16bf ldfrag(const bf16* p) {
  union { v16bf v; uint4 q[2]; } u;
  u.q[0] = *(const uint4*)(p);
  u.q[1] = *(const uint4*)(p + 16);
  return u.v;
}

__device__ __forceinline__ v8f wmma_bf16(v16bf a, v16bf b, v8f c) {
  return __builtin_amdgcn_wmma_f32_16x16x32_bf16(false, a, false, b,
                                                 (short)0, c, false, false);
}

// 16x16 tile of A(128x512,row-major bf16) x B^T(rows=N,cols=K=512, bf16).
// All 16 A-fragments are batch-loaded first (one load clause, ~1 L2 latency
// on the critical path); B^T fragments are t-loop-invariant for Q^T/R^T and
// get register-hoisted by the compiler.
__device__ __forceinline__ v8f mm512(const bf16* __restrict__ A,
                                     const bf16* __restrict__ BT,
                                     int m0, int n0, int lane) {
  const int lo   = lane & 15;
  const int koff = (lane >> 4) << 3;
  const bf16* ap = A  + (size_t)(m0 + lo) * 512 + koff;
  const bf16* bp = BT + (size_t)(n0 + lo) * 512 + koff;
  v16bf af[16];
#pragma unroll
  for (int i = 0; i < 16; ++i) af[i] = ldfrag(ap + 32 * i);
  v8f acc = {};
#pragma unroll
  for (int i = 0; i < 16; ++i) {
    v16bf b = ldfrag(bp + 32 * i);
    acc = wmma_bf16(af[i], b, acc);
  }
  return acc;
}

// Device-wide barrier: monotonic counter in ws, reset by init kernel each
// launch. Release fence by all threads before arrival, acquire after.
__device__ __forceinline__ void gsync(unsigned* bar, unsigned target) {
  __threadfence();                      // release: make this thread's stores visible
  __syncthreads();
  if (threadIdx.x == 0) {
    __hip_atomic_fetch_add(bar, 1u, __ATOMIC_RELEASE, __HIP_MEMORY_SCOPE_AGENT);
    while (__hip_atomic_load(bar, __ATOMIC_ACQUIRE, __HIP_MEMORY_SCOPE_AGENT) < target) {
      __builtin_amdgcn_s_sleep(1);
    }
  }
  __syncthreads();
  __threadfence();                      // acquire: discard stale cached lines
}

// ---------- init: bf16 transposed weights, zero state, fused bias, barrier reset ----------

__global__ __launch_bounds__(TPB) void mog_init(
    const float* __restrict__ Wih, const float* __restrict__ Whh,
    const float* __restrict__ Q,   const float* __restrict__ R,
    const float* __restrict__ bih, const float* __restrict__ bhh,
    bf16* __restrict__ WihT, bf16* __restrict__ WhhT,
    bf16* __restrict__ QT,   bf16* __restrict__ RT,
    float* __restrict__ bsum,
    float* __restrict__ ht_f, bf16* __restrict__ ht_b, float* __restrict__ Ct,
    unsigned* __restrict__ bar) {
  int i = blockIdx.x * blockDim.x + threadIdx.x;
  if (i < 2048 * 512) {                 // W^T[n][k] = W[k][n], (4H x D)
    int n = i / 512, k = i % 512;
    WihT[i] = to_bf16(Wih[(size_t)k * 2048 + n]);
    WhhT[i] = to_bf16(Whh[(size_t)k * 2048 + n]);
  }
  if (i < 512 * 512) {                  // Q^T, R^T (512 x 512)
    int n = i / 512, k = i % 512;
    QT[i] = to_bf16(Q[(size_t)k * 512 + n]);
    RT[i] = to_bf16(R[(size_t)k * 512 + n]);
  }
  if (i < B_ * H_) { ht_f[i] = 0.0f; ht_b[i] = to_bf16(0.0f); Ct[i] = 0.0f; }
  if (i < 4 * H_)  { bsum[i] = bih[i] + bhh[i]; }
  if (i == 0)      { *bar = 0u; }
}

// ---------- persistent Mogrifier-LSTM ----------

__global__ __launch_bounds__(TPB) void mog_lstm(
    const float* __restrict__ x,      // (B,S,D) f32
    const float* __restrict__ bsum,   // (4H) f32  (bih+bhh)
    float* __restrict__ out,          // hidden_seq (B,S,H) ++ ht (B,H) ++ Ct (B,H)
    const bf16* __restrict__ WihT, const bf16* __restrict__ WhhT,
    const bf16* __restrict__ QT,   const bf16* __restrict__ RT,
    float* __restrict__ xt_f, bf16* __restrict__ xt_b,
    float* __restrict__ ht_f, bf16* __restrict__ ht_b,
    float* __restrict__ Ct,
    unsigned* __restrict__ bar) {
  const int lane = threadIdx.x & 31;
  const int wave = (blockIdx.x << 3) + (threadIdx.x >> 5);  // 0..255
  const int m0 = (wave >> 5) << 4;                          // 8 row tiles (B=128)
  const int n0 = (wave & 31) << 4;                          // 32 col tiles (512)
  const int lo = lane & 15, hi = lane >> 4;
  const int koff = hi << 3;
  const size_t BSH = (size_t)B_ * S_ * H_;
  unsigned tgt = 0;

  for (int t = 0; t < S_; ++t) {
    // ---- mogrify i=1: U = ht @ Q ; xt = 2*sig(U) * x[:,t,:]
    {
      v8f u = mm512(ht_b, QT, m0, n0, lane);
#pragma unroll
      for (int r = 0; r < 8; ++r) {
        int b = m0 + r + (hi << 3), d = n0 + lo;
        float ox = x[(size_t)b * ((size_t)S_ * D_) + (size_t)t * D_ + d];
        float nx = 2.0f * sigmoid_(u[r]) * ox;
        xt_f[b * D_ + d] = nx;
        xt_b[b * D_ + d] = to_bf16(nx);
      }
    }
    tgt += NBLK; gsync(bar, tgt);

    // ---- mogrify i=2: V = xt @ R ; ht = 2*sig(V)*ht
    {
      v8f v = mm512(xt_b, RT, m0, n0, lane);
#pragma unroll
      for (int r = 0; r < 8; ++r) {
        int b = m0 + r + (hi << 3), h = n0 + lo;
        float nh = 2.0f * sigmoid_(v[r]) * ht_f[b * H_ + h];
        ht_f[b * H_ + h] = nh;
        ht_b[b * H_ + h] = to_bf16(nh);
      }
    }
    tgt += NBLK; gsync(bar, tgt);

    // ---- mogrify i=3,5 (odd) and i=4 (even)
#pragma unroll 1
    for (int it = 0; it < 2; ++it) {
      {   // odd: xt = 2*sig(ht@Q)*xt
        v8f u = mm512(ht_b, QT, m0, n0, lane);
#pragma unroll
        for (int r = 0; r < 8; ++r) {
          int b = m0 + r + (hi << 3), d = n0 + lo;
          float nx = 2.0f * sigmoid_(u[r]) * xt_f[b * D_ + d];
          xt_f[b * D_ + d] = nx;
          xt_b[b * D_ + d] = to_bf16(nx);
        }
      }
      tgt += NBLK; gsync(bar, tgt);
      if (it == 0) {  // even (i=4): ht = 2*sig(xt@R)*ht
        v8f v = mm512(xt_b, RT, m0, n0, lane);
#pragma unroll
        for (int r = 0; r < 8; ++r) {
          int b = m0 + r + (hi << 3), h = n0 + lo;
          float nh = 2.0f * sigmoid_(v[r]) * ht_f[b * H_ + h];
          ht_f[b * H_ + h] = nh;
          ht_b[b * H_ + h] = to_bf16(nh);
        }
        tgt += NBLK; gsync(bar, tgt);
      }
    }

    // ---- gates: G = xt@Wih + ht@Whh (+bias). One K-loop, all 4 quadrants
    //      inside it -> 10 independent load streams per K-step so weight
    //      loads overlap with WMMAs. Fused LSTM cell update on the 16x16
    //      (b,h) tile.
    {
      v8f acc[4] = {v8f{}, v8f{}, v8f{}, v8f{}};
      const bf16* apx = xt_b + (size_t)(m0 + lo) * 512 + koff;
      const bf16* aph = ht_b + (size_t)(m0 + lo) * 512 + koff;
      const bf16* bpi = WihT + (size_t)(n0 + lo) * 512 + koff;   // +q*512*512
      const bf16* bph = WhhT + (size_t)(n0 + lo) * 512 + koff;
#pragma unroll
      for (int k = 0; k < 512; k += 32) {
        v16bf ax = ldfrag(apx + k);
        v16bf ah = ldfrag(aph + k);
#pragma unroll
        for (int q = 0; q < 4; ++q) {
          v16bf bi = ldfrag(bpi + (size_t)q * (512 * 512) + k);
          v16bf bh = ldfrag(bph + (size_t)q * (512 * 512) + k);
          acc[q] = wmma_bf16(ax, bi, acc[q]);
          acc[q] = wmma_bf16(ah, bh, acc[q]);
        }
      }
#pragma unroll
      for (int r = 0; r < 8; ++r) {
        int b = m0 + r + (hi << 3), h = n0 + lo;
        float gi = acc[0][r] + bsum[h];
        float gf = acc[1][r] + bsum[512 + h];
        float gg = acc[2][r] + bsum[1024 + h];
        float go = acc[3][r] + bsum[1536 + h];
        float itg = sigmoid_(gi), ftg = sigmoid_(gf);
        float gtg = tanh_(gg),    otg = sigmoid_(go);
        float c  = ftg * Ct[b * H_ + h] + itg * gtg;
        Ct[b * H_ + h] = c;
        float hn = otg * tanh_(c);
        ht_f[b * H_ + h] = hn;
        ht_b[b * H_ + h] = to_bf16(hn);
        out[(size_t)b * ((size_t)S_ * H_) + (size_t)t * H_ + h] = hn;
        if (t == S_ - 1) {
          out[BSH + (size_t)b * H_ + h] = hn;                       // final ht
          out[BSH + (size_t)B_ * H_ + (size_t)b * H_ + h] = c;      // final Ct
        }
      }
    }
    tgt += NBLK; gsync(bar, tgt);
  }
}

// ---------- host ----------

extern "C" void kernel_launch(void* const* d_in, const int* in_sizes, int n_in,
                              void* d_out, int out_size, void* d_ws, size_t ws_size,
                              hipStream_t stream) {
  (void)in_sizes; (void)n_in; (void)out_size; (void)ws_size;
  const float* x   = (const float*)d_in[0];
  const float* Wih = (const float*)d_in[1];
  const float* Whh = (const float*)d_in[2];
  const float* bih = (const float*)d_in[3];
  const float* bhh = (const float*)d_in[4];
  const float* Q   = (const float*)d_in[5];
  const float* R   = (const float*)d_in[6];
  float* out = (float*)d_out;

  uint8_t* ws = (uint8_t*)d_ws;
  size_t off = 0;
  auto take = [&](size_t bytes) -> void* {
    void* p = ws + off;
    off += (bytes + 255) & ~(size_t)255;
    return p;
  };
  unsigned* bar  = (unsigned*)take(sizeof(unsigned));
  bf16* WihT = (bf16*)take(2048ull * 512 * sizeof(bf16));
  bf16* WhhT = (bf16*)take(2048ull * 512 * sizeof(bf16));
  bf16* QT   = (bf16*)take(512ull * 512 * sizeof(bf16));
  bf16* RT   = (bf16*)take(512ull * 512 * sizeof(bf16));
  float* bsum = (float*)take(2048ull * sizeof(float));
  float* xt_f = (float*)take((size_t)B_ * D_ * sizeof(float));
  bf16*  xt_b = (bf16*) take((size_t)B_ * D_ * sizeof(bf16));
  float* ht_f = (float*)take((size_t)B_ * H_ * sizeof(float));
  bf16*  ht_b = (bf16*) take((size_t)B_ * H_ * sizeof(bf16));
  float* Ct   = (float*)take((size_t)B_ * H_ * sizeof(float));

  // transpose/convert weights, zero state, reset barrier
  mog_init<<<(2048 * 512) / TPB, TPB, 0, stream>>>(
      Wih, Whh, Q, R, bih, bhh, WihT, WhhT, QT, RT, bsum, ht_f, ht_b, Ct, bar);

  // persistent cooperative recurrence: 32 blocks x 256 thr = 256 waves,
  // one 16x16 WMMA tile per wave per phase.
  mog_lstm<<<NBLK, TPB, 0, stream>>>(
      x, bsum, out, WihT, WhhT, QT, RT, xt_f, xt_b, ht_f, ht_b, Ct, bar);
}
